// SSMS4_19430432047161
// MI455X (gfx1250) — matrix operand order repaired and checked by reading, not compiled, not run on hardware
//
#include <hip/hip_runtime.h>

typedef __attribute__((ext_vector_type(16))) _Float16 v16h;
typedef __attribute__((ext_vector_type(8)))  float    v8f;

#define L_LEN 8192
#define CH    512
#define NS    64          // state dim
#define TC    64          // chunk (tile) length in time
#define NJ    (L_LEN/TC)  // 128 chunks
#define SRATE 4096.0f

// ---------------------------------------------------------------------------
// Workspace layout (d_ws):
//   [0      , 8KB ) : Mh  f16 [64][64]  intra-chunk kernel matrix, M[u][v]=k[u-v] (lower-tri)
//   [8KB    , 16KB) : Dh  f16 [64][64]  state-injection matrix, D[n][v]=Bbar_n*Abar_n^(T-1-v)
//   [16KB   , 24KB) : Eh  f16 [64][64]  state-readout matrix,   E[u][n]=C_n*Abar_n^(u+1)
//   [24KB   , +256) : decayT f32 [64]   Abar_n^T
//   [32KB   , +16MB): H   f32 [J][N][C] chunk states (H_in, then in-place h_start)
// ---------------------------------------------------------------------------

// ---- prep: build the small f16 operand matrices (1 block, 64 threads) -----
__global__ void __launch_bounds__(64)
ssm_prep(const float* __restrict__ logA, const float* __restrict__ Bp,
         const float* __restrict__ Cp,
         _Float16* __restrict__ Mh, _Float16* __restrict__ Dh,
         _Float16* __restrict__ Eh, float* __restrict__ decayT)
{
    __shared__ float sAbar[NS], sW[NS], sK[TC];
    const int n = threadIdx.x;                  // 0..63

    const float A    = -__expf(logA[n]);
    const float Abar = __expf(A / SRATE);
    const float Bbar = (Abar - 1.0f) * Bp[n] / A;
    const float Cn   = Cp[n];

    sAbar[n]  = Abar;
    sW[n]     = Cn * Bbar;                      // w_n = C_n * Bbar_n
    decayT[n] = powf(Abar, (float)TC);

    for (int v = 0; v < TC; ++v)
        Dh[n * TC + v] = (_Float16)(Bbar * powf(Abar, (float)(TC - 1 - v)));
    for (int u = 0; u < TC; ++u)
        Eh[u * NS + n] = (_Float16)(Cn * powf(Abar, (float)(u + 1)));
    __syncthreads();

    // k[d] = sum_n w_n * Abar_n^d   (thread d computes one lag)
    float kd = 0.0f;
    for (int i = 0; i < NS; ++i)
        kd += sW[i] * powf(sAbar[i], (float)n);
    sK[n] = kd;
    __syncthreads();

    // M[u][v] = k[u-v] for v<=u else 0  (thread handles row u = n)
    for (int v = 0; v < TC; ++v)
        Mh[n * TC + v] = (n >= v) ? (_Float16)sK[n - v] : (_Float16)0.0f;
}

// ---- fragment loaders ------------------------------------------------------
// 16-bit A-matrix (16xK) layout: per lane (m=lane&15, hi=lane>>4), within a
// 32-wide K block the 16 halves are one contiguous run (matches B fragments):
//   a[0..7]  = K = kb*32 + 16*hi     + 0..7
//   a[8..15] = K = kb*32 + 16*hi + 8 + 0..7
__device__ __forceinline__ v16h load_a_frag(const _Float16* __restrict__ row,
                                            int kb, int hi)
{
    const _Float16* p0 = row + kb * 32 + 16 * hi;
    const _Float16* p1 = p0 + 8;
    v16h a;
#pragma unroll
    for (int i = 0; i < 8; ++i) { a[i] = p0[i]; a[8 + i] = p1[i]; }
    return a;
}

#define WMMA_F16(acc, a, b)                                                   \
    acc = __builtin_amdgcn_wmma_f32_16x16x32_f16(false, (a), false, (b),      \
                                                 (short)0, (acc), false, false)

// ---- pass 1: H_in[j] = D @ X_j  (one wave per (chunk, 16-ch tile)) ---------
__global__ void __launch_bounds__(32)
ssm_chunk_states(const float* __restrict__ x, const _Float16* __restrict__ Dh,
                 float* __restrict__ H)
{
    const int j    = blockIdx.x >> 5;       // chunk
    const int c0   = (blockIdx.x & 31) * 16;// channel tile base
    const int lane = threadIdx.x;
    const int m    = lane & 15;
    const int hi   = lane >> 4;

    // Stage X_j transposed: tile[chan][time] so B-fragments are contiguous.
    __shared__ __align__(32) _Float16 tile[16][TC];
    for (int idx = lane; idx < 16 * TC; idx += 32) {
        const int t = idx >> 4, c = idx & 15;
        tile[c][t] = (_Float16)x[(size_t)(j * TC + t) * CH + c0 + c];
    }
    __syncthreads();

    v8f a0 = {}, a1 = {}, a2 = {}, a3 = {};
#pragma unroll
    for (int kb = 0; kb < 2; ++kb) {
        // B 32x16 layout: lane n=m, K-half=hi -> 16 contiguous halves in LDS
        const v16h b = *(const v16h*)&tile[m][kb * 32 + hi * 16];
        v16h a;
        a = load_a_frag(Dh + (size_t)( 0 + m) * TC, kb, hi); WMMA_F16(a0, a, b);
        a = load_a_frag(Dh + (size_t)(16 + m) * TC, kb, hi); WMMA_F16(a1, a, b);
        a = load_a_frag(Dh + (size_t)(32 + m) * TC, kb, hi); WMMA_F16(a2, a, b);
        a = load_a_frag(Dh + (size_t)(48 + m) * TC, kb, hi); WMMA_F16(a3, a, b);
    }

    // C/D f32 layout: VGPR r -> row r + 8*hi, col = m
    float* Hj = H + (size_t)j * NS * CH + c0 + m;
#pragma unroll
    for (int r = 0; r < 8; ++r) {
        const int row = r + 8 * hi;
        Hj[(size_t)( 0 + row) * CH] = a0[r];
        Hj[(size_t)(16 + row) * CH] = a1[r];
        Hj[(size_t)(32 + row) * CH] = a2[r];
        Hj[(size_t)(48 + row) * CH] = a3[r];
    }
}

// ---- pass 2: cross-chunk scan, in place: H[j] := h_start[j] ----------------
__global__ void __launch_bounds__(256)
ssm_scan(float* __restrict__ H, const float* __restrict__ decayT)
{
    const int idx = blockIdx.x * 256 + threadIdx.x; // idx = n*CH + c
    const float dT = decayT[idx / CH];
    float h = 0.0f;
    float* p = H + idx;
    for (int j = 0; j < NJ; ++j) {
        const float v = p[(size_t)j * NS * CH];
        p[(size_t)j * NS * CH] = h;                 // h at chunk start
        h = dT * h + v;
    }
}

// ---- pass 3: Y_j = M @ X_j + E @ h_start[j] --------------------------------
__global__ void __launch_bounds__(32)
ssm_output(const float* __restrict__ x, const float* __restrict__ H,
           const _Float16* __restrict__ Mh, const _Float16* __restrict__ Eh,
           float* __restrict__ y)
{
    const int j    = blockIdx.x >> 5;
    const int c0   = (blockIdx.x & 31) * 16;
    const int lane = threadIdx.x;
    const int m    = lane & 15;
    const int hi   = lane >> 4;

    __shared__ __align__(32) _Float16 xt[16][TC];   // [chan][time]
    __shared__ __align__(32) _Float16 ht[16][NS];   // [chan][state]
    for (int idx = lane; idx < 16 * TC; idx += 32) {
        const int t = idx >> 4, c = idx & 15;       // t doubles as state index
        xt[c][t] = (_Float16)x[(size_t)(j * TC + t) * CH + c0 + c];
        ht[c][t] = (_Float16)H[(size_t)j * NS * CH + (size_t)t * CH + c0 + c];
    }
    __syncthreads();

    v8f a0 = {}, a1 = {}, a2 = {}, a3 = {};
#pragma unroll
    for (int kb = 0; kb < 2; ++kb) {
        const v16h bx = *(const v16h*)&xt[m][kb * 32 + hi * 16];
        const v16h bh = *(const v16h*)&ht[m][kb * 32 + hi * 16];
        v16h a;
        a = load_a_frag(Mh + (size_t)( 0 + m) * TC, kb, hi); WMMA_F16(a0, a, bx);
        a = load_a_frag(Eh + (size_t)( 0 + m) * NS, kb, hi); WMMA_F16(a0, a, bh);
        a = load_a_frag(Mh + (size_t)(16 + m) * TC, kb, hi); WMMA_F16(a1, a, bx);
        a = load_a_frag(Eh + (size_t)(16 + m) * NS, kb, hi); WMMA_F16(a1, a, bh);
        a = load_a_frag(Mh + (size_t)(32 + m) * TC, kb, hi); WMMA_F16(a2, a, bx);
        a = load_a_frag(Eh + (size_t)(32 + m) * NS, kb, hi); WMMA_F16(a2, a, bh);
        a = load_a_frag(Mh + (size_t)(48 + m) * TC, kb, hi); WMMA_F16(a3, a, bx);
        a = load_a_frag(Eh + (size_t)(48 + m) * NS, kb, hi); WMMA_F16(a3, a, bh);
    }

    float* yj = y + (size_t)j * TC * CH + c0 + m;
#pragma unroll
    for (int r = 0; r < 8; ++r) {
        const int row = r + 8 * hi;                 // time index within chunk
        yj[(size_t)( 0 + row) * CH] = a0[r];
        yj[(size_t)(16 + row) * CH] = a1[r];
        yj[(size_t)(32 + row) * CH] = a2[r];
        yj[(size_t)(48 + row) * CH] = a3[r];
    }
}

// ---------------------------------------------------------------------------
extern "C" void kernel_launch(void* const* d_in, const int* in_sizes, int n_in,
                              void* d_out, int out_size, void* d_ws, size_t ws_size,
                              hipStream_t stream)
{
    (void)in_sizes; (void)n_in; (void)out_size; (void)ws_size;

    const float* x    = (const float*)d_in[0];  // (L, C)
    const float* logA = (const float*)d_in[1];  // (N,)
    const float* Bp   = (const float*)d_in[2];  // (N,)
    const float* Cp   = (const float*)d_in[3];  // (N,)
    float*       y    = (float*)d_out;          // (L, C)

    char* ws = (char*)d_ws;
    _Float16* Mh     = (_Float16*)(ws);
    _Float16* Dh     = (_Float16*)(ws + 8 * 1024);
    _Float16* Eh     = (_Float16*)(ws + 16 * 1024);
    float*    decayT = (float*)(ws + 24 * 1024);
    float*    H      = (float*)(ws + 32 * 1024); // J*N*C f32 = 16 MB

    ssm_prep<<<1, 64, 0, stream>>>(logA, Bp, Cp, Mh, Dh, Eh, decayT);
    ssm_chunk_states<<<NJ * 32, 32, 0, stream>>>(x, Dh, H);
    ssm_scan<<<(NS * CH) / 256, 256, 0, stream>>>(H, decayT);
    ssm_output<<<NJ * 32, 32, 0, stream>>>(x, H, Mh, Eh, y);
}
